// MemoryLayer_36112085025326
// MI455X (gfx1250) — matrix operand order repaired and checked by reading, not compiled, not run on hardware
//
#include <hip/hip_runtime.h>
#include <math.h>

// ---------------------------------------------------------------------------
// MemoryLayer for MI455X (gfx1250): sequential top-k softmax recurrence with a
// WMMA-bf16 GEMM (omega @ values^T) every 16 steps. All heavy state (alpha
// 2MB, omega tile 16MB, values-bf16 32MB) is L2-resident on a 192MB L2 chip.
// Launch-count optimized: one kernel per recurrence step (redundant in-block
// combine of the previous step's 64 partials + ping-pong partial buffers).
// ---------------------------------------------------------------------------

#define DD      128
#define NN      131072
#define B_SZ    4
#define TOPK    128
#define TINV    10.0f                  // 1 / TEMP
#define LN_EPS  1e-5f
#define NBLK    64                     // reduction blocks per batch over N
#define CHUNK   (NN / NBLK)            // 2048 elements per block
#define SIMS_SCALE 0.08838834764831845f // 1/sqrt(128)

typedef __attribute__((ext_vector_type(16))) __bf16 v16bf;
typedef __attribute__((ext_vector_type(8)))  __bf16 v8bf;
typedef __attribute__((ext_vector_type(8)))  float  v8f;

__device__ __forceinline__ unsigned short f2bf(float f) {
  unsigned u = __float_as_uint(f);
  u += 0x7FFFu + ((u >> 16) & 1u);          // round-to-nearest-even
  return (unsigned short)(u >> 16);
}

// merge two (max m, s = sum exp((x-m)*sc)) online-softmax partials
__device__ __forceinline__ void mergeMS(float& m, float& s, float m2, float s2, float sc) {
  if (m2 > m) { s = s * __expf((m - m2) * sc) + s2; m = m2; }
  else        { s = s + s2 * __expf((m2 - m) * sc); }
}

__device__ __forceinline__ void blockReduceMS(float m, float s, float sc,
                                              float* sm, float* ss, float2* dst) {
  int tid = threadIdx.x;
  sm[tid] = m; ss[tid] = s;
  __syncthreads();
  for (int w = 128; w > 0; w >>= 1) {
    if (tid < w) mergeMS(sm[tid], ss[tid], sm[tid + w], ss[tid + w], sc);
    __syncthreads();
  }
  if (tid == 0) *dst = make_float2(sm[0], ss[0]);
}

// redundant (per-thread) combine of NBLK partials -> (M, 1/S); tiny & L2-hot
__device__ __forceinline__ void combineAll(const float2* __restrict__ part, int b,
                                           float sc, float& M, float& invS) {
  float m = -INFINITY, s = 0.f;
  for (int i = 0; i < NBLK; ++i) {
    float2 p = part[b * NBLK + i];
    mergeMS(m, s, p.x, p.y, sc);
  }
  M = m; invS = 1.0f / s;
}

// q[b][d] = (mean_t x[b][t][:]) @ W_q
__global__ void k_qproj(const float* __restrict__ x, const float* __restrict__ Wq,
                        float* __restrict__ q) {
  __shared__ float xm[DD];
  int b = blockIdx.x, d = threadIdx.x;
  const float* xb = x + (size_t)b * DD * DD;
  float s = 0.f;
  for (int t = 0; t < DD; ++t) s += xb[t * DD + d];
  xm[d] = s * (1.0f / DD);
  __syncthreads();
  float acc = 0.f;
  for (int e = 0; e < DD; ++e) acc += xm[e] * Wq[e * DD + d];
  q[b * DD + d] = acc;
}

// values f32 -> bf16 (done once per launch; 32MB output stays in L2)
__global__ void k_cvt(const float* __restrict__ v, unsigned short* __restrict__ o) {
  size_t i = ((size_t)blockIdx.x * 256 + threadIdx.x) * 8;
#pragma unroll
  for (int j = 0; j < 8; ++j) o[i + j] = f2bf(v[i + j]);
}

// sims[b][n] = -(q.keys_col)/sqrt(128); emit softmax partials (scale 1)
__global__ void k_sims(const float* __restrict__ keys, const float* __restrict__ q,
                       float* __restrict__ alpha, float2* __restrict__ simspart) {
  __shared__ float qs[DD];
  __shared__ float sm[256], ss[256];
  int b = blockIdx.y, tid = threadIdx.x;
  if (tid < DD) qs[tid] = q[b * DD + tid];
  __syncthreads();
  int nbase = blockIdx.x * CHUNK;
  float acc[8];
#pragma unroll
  for (int i = 0; i < 8; ++i) acc[i] = 0.f;
  for (int d = 0; d < DD; ++d) {
    float qd = qs[d];
    const float* kr = keys + (size_t)d * NN + nbase + tid;
#pragma unroll
    for (int i = 0; i < 8; ++i) acc[i] += qd * kr[i * 256];
  }
  float m = -INFINITY, s = 0.f;
#pragma unroll
  for (int i = 0; i < 8; ++i) {
    float v = -acc[i] * SIMS_SCALE;
    alpha[(size_t)b * NN + nbase + tid + i * 256] = v;
    mergeMS(m, s, v, 1.0f, 1.0f);
  }
  blockReduceMS(m, s, 1.0f, sm, ss, &simspart[b * NBLK + blockIdx.x]);
}

// alpha0 = softmax(sims) (combining sims partials in-block), emit step-0
// partials (scale TINV) into pout.
__global__ void k_alpha0(float* __restrict__ alpha, const float2* __restrict__ simspart,
                         float2* __restrict__ pout) {
  __shared__ float sm[256], ss[256];
  int b = blockIdx.y, tid = threadIdx.x;
  float M, inv;
  combineAll(simspart, b, 1.0f, M, inv);
  int nbase = blockIdx.x * CHUNK;
  float m = -INFINITY, s = 0.f;
#pragma unroll
  for (int i = 0; i < 8; ++i) {
    size_t idx = (size_t)b * NN + nbase + tid + i * 256;
    float a0 = __expf(alpha[idx] - M) * inv;
    alpha[idx] = a0;
    mergeMS(m, s, a0, 1.0f, TINV);
  }
  blockReduceMS(m, s, TINV, sm, ss, &pout[b * NBLK + blockIdx.x]);
}

// one recurrence step (single kernel): combine previous partials in-block,
// om = softmax(alpha/t); store bf16 omega tile; alpha += log1p(-om);
// emit next-step partials into the ping-pong output buffer.
__global__ void k_step(float* __restrict__ alpha, const float2* __restrict__ pin,
                       float2* __restrict__ pout, unsigned short* __restrict__ omega,
                       int kk) {
  __shared__ float sm[256], ss[256];
  int b = blockIdx.y, tid = threadIdx.x;
  float M, inv;
  combineAll(pin, b, TINV, M, inv);
  int nbase = blockIdx.x * CHUNK;
  float m = -INFINITY, s = 0.f;
#pragma unroll
  for (int i = 0; i < 8; ++i) {
    size_t idx = (size_t)b * NN + nbase + tid + i * 256;
    float a = alpha[idx];
    float om = __expf((a - M) * TINV) * inv;
    omega[((size_t)kk * B_SZ + b) * NN + nbase + tid + i * 256] = f2bf(om);
    float an = a + log1pf(-om);
    alpha[idx] = an;
    mergeMS(m, s, an, 1.0f, TINV);
  }
  blockReduceMS(m, s, TINV, sm, ss, &pout[b * NBLK + blockIdx.x]);
}

// WMMA GEMM: partial knear[b][m=16 k-rows][f=128] over one 16-step omega tile.
// A (16x32 bf16): lane L holds row M=L&15; halves 0..7 -> K=kb..kb+7,
//                 halves 8..15 -> K=kb+16..kb+23 with kb = (L>=16)?8:0.
// B (32x16 bf16): lane L holds col N=L&15; same K order as A (kb=(L>=16)?8:0).
// Split-K into 64 deterministic partial tiles (no float atomics).
__global__ void __launch_bounds__(128) k_gemm(const unsigned short* __restrict__ omega,
                                              const unsigned short* __restrict__ vals,
                                              float* __restrict__ parts) {
  int b = blockIdx.y;
  int tid = threadIdx.x;
  int wave = tid >> 5, lane = tid & 31;
  int row = lane & 15, hi = lane >> 4;
  int split = blockIdx.x * 4 + wave;              // 0..63
  int nbase = split * (NN / 64);                  // 2048-wide contraction chunk

  const __bf16* om = (const __bf16*)omega;
  const __bf16* vb = (const __bf16*)vals;

  v8f zero = {0.f, 0.f, 0.f, 0.f, 0.f, 0.f, 0.f, 0.f};
  v8f acc[8];
#pragma unroll
  for (int t = 0; t < 8; ++t) acc[t] = zero;

  const __bf16* pa  = om + ((size_t)row * B_SZ + b) * NN + nbase + hi * 8;
  const __bf16* pv0 = vb + (size_t)row * NN + nbase + hi * 8;

  for (int n = 0; n < NN / 64; n += 32) {
    __builtin_prefetch(pa + n + 128, 0, 1);
    v8bf alo = *(const v8bf*)(pa + n);
    v8bf ahi = *(const v8bf*)(pa + n + 16);
    v16bf A = __builtin_shufflevector(alo, ahi, 0, 1, 2, 3, 4, 5, 6, 7,
                                      8, 9, 10, 11, 12, 13, 14, 15);
#pragma unroll
    for (int t = 0; t < 8; ++t) {
      const __bf16* pb = pv0 + (size_t)t * 16 * NN + n;
      v8bf b0 = *(const v8bf*)(pb);
      v8bf b1 = *(const v8bf*)(pb + 16);
      v16bf Bv = __builtin_shufflevector(b0, b1, 0, 1, 2, 3, 4, 5, 6, 7,
                                         8, 9, 10, 11, 12, 13, 14, 15);
      acc[t] = __builtin_amdgcn_wmma_f32_16x16x32_bf16(
          false, A, false, Bv, (short)0, acc[t], false, false);
    }
  }

  // D layout: lane 0-15 -> N=lane, VGPR r -> M=r; lanes 16-31 -> M=8+r.
  float* outp = parts + ((size_t)split * B_SZ + b) * 16 * 128;
#pragma unroll
  for (int t = 0; t < 8; ++t) {
#pragma unroll
    for (int r = 0; r < 8; ++r) {
      int mrow = hi * 8 + r;
      int f = t * 16 + row;
      outp[mrow * 128 + f] = acc[t][r];
    }
  }
}

// single deterministic split-K reduction over all 8 groups at the end:
// knear[b][k][f] = sum_{split} parts[group(k)][split][b][k&15][f]
__global__ void k_reduce_kn(const float* __restrict__ parts, float* __restrict__ knear) {
  int idx = blockIdx.x * 256 + threadIdx.x;       // 0..65535 = (b,k,f)
  int f = idx & 127;
  int k = (idx >> 7) & 127;
  int b = idx >> 14;
  int g = k >> 4, m = k & 15;
  const float* p = parts + (size_t)g * 64 * (B_SZ * 16 * 128)
                         + ((size_t)b * 16 + m) * 128 + f;
  float s = 0.f;
  for (int sp = 0; sp < 64; ++sp) s += p[(size_t)sp * (B_SZ * 16 * 128)];
  knear[idx] = s;
}

// cast1[b][k][v] = sum_u knear[b][k][u] * W_ffdim[v][u]
__global__ void k_ffdim(const float* __restrict__ knear, const float* __restrict__ Wd,
                        float* __restrict__ cast1) {
  int b = blockIdx.y;
  int o = blockIdx.x * 256 + threadIdx.x;         // 0..16383
  int k = o >> 7, v = o & 127;
  const float* kr = knear + ((size_t)b * 128 + k) * 128;
  const float* wr = Wd + v * 128;
  float s = 0.f;
  for (int u = 0; u < 128; ++u) s += kr[u] * wr[u];
  cast1[((size_t)b * 128 + k) * 128 + v] = s;
}

// y[b][t][j] = x[b][t][j] + sum_k cast1[b][k][j]*W_ffctx[t][k], then LayerNorm(j)
__global__ void k_final(const float* __restrict__ x, const float* __restrict__ cast1,
                        const float* __restrict__ Wc, const float* __restrict__ gamma,
                        const float* __restrict__ beta, float* __restrict__ out) {
  __shared__ float r1[128], r2[128];
  int b = blockIdx.y, t = blockIdx.x, j = threadIdx.x;
  float s = 0.f;
  for (int k = 0; k < 128; ++k)
    s += cast1[((size_t)b * 128 + k) * 128 + j] * Wc[t * 128 + k];
  float y = x[((size_t)b * 128 + t) * 128 + j] + s;
  r1[j] = y; r2[j] = y * y;
  __syncthreads();
  for (int w = 64; w > 0; w >>= 1) {
    if (j < w) { r1[j] += r1[j + w]; r2[j] += r2[j + w]; }
    __syncthreads();
  }
  float mu  = r1[0] * (1.0f / 128.0f);
  float var = r2[0] * (1.0f / 128.0f) - mu * mu;
  out[((size_t)b * 128 + t) * 128 + j] =
      (y - mu) * rsqrtf(var + LN_EPS) * gamma[j] + beta[j];
}

extern "C" void kernel_launch(void* const* d_in, const int* in_sizes, int n_in,
                              void* d_out, int out_size, void* d_ws, size_t ws_size,
                              hipStream_t stream) {
  (void)in_sizes; (void)n_in; (void)out_size; (void)ws_size;
  const float* x      = (const float*)d_in[0];
  const float* keys   = (const float*)d_in[1];
  const float* values = (const float*)d_in[2];
  const float* Wq     = (const float*)d_in[3];
  const float* Wd     = (const float*)d_in[4];
  const float* Wc     = (const float*)d_in[5];
  const float* gamma  = (const float*)d_in[6];
  const float* beta   = (const float*)d_in[7];
  float* out = (float*)d_out;

  // workspace carve (~67 MB total), 256B aligned
  char* ws = (char*)d_ws;
  size_t off = 0;
  auto carve = [&](size_t bytes) -> char* {
    char* p = ws + off;
    off += (bytes + 255) & ~(size_t)255;
    return p;
  };
  float*          alpha   = (float*)carve((size_t)B_SZ * NN * 4);             // 2 MB
  float2*         simspart= (float2*)carve((size_t)B_SZ * NBLK * 8);
  float2*         partA   = (float2*)carve((size_t)B_SZ * NBLK * 8);          // ping
  float2*         partB   = (float2*)carve((size_t)B_SZ * NBLK * 8);          // pong
  float*          q       = (float*)carve((size_t)B_SZ * DD * 4);
  float*          knear   = (float*)carve((size_t)B_SZ * 128 * 128 * 4);      // 256 KB
  float*          cast1   = (float*)carve((size_t)B_SZ * 128 * 128 * 4);      // 256 KB
  float*          parts   = (float*)carve((size_t)8 * 64 * B_SZ * 16 * 128 * 4); // 16 MB
  unsigned short* omega   = (unsigned short*)carve((size_t)16 * B_SZ * NN * 2);  // 16 MB
  unsigned short* valsb   = (unsigned short*)carve((size_t)DD * NN * 2);      // 32 MB

  float2* pp[2] = { partA, partB };

  k_qproj<<<B_SZ, DD, 0, stream>>>(x, Wq, q);
  k_cvt<<<(DD * NN) / (256 * 8), 256, 0, stream>>>(values, valsb);
  k_sims<<<dim3(NBLK, B_SZ), 256, 0, stream>>>(keys, q, alpha, simspart);
  k_alpha0<<<dim3(NBLK, B_SZ), 256, 0, stream>>>(alpha, simspart, pp[0]);

  for (int k = 0; k < TOPK; ++k) {
    // step k reads pp[k&1], writes pp[(k+1)&1] (ping-pong avoids RAW across blocks)
    k_step<<<dim3(NBLK, B_SZ), 256, 0, stream>>>(alpha, pp[k & 1], pp[(k + 1) & 1],
                                                 omega, k & 15);
    if ((k & 15) == 15) {
      k_gemm<<<dim3(16, B_SZ), 128, 0, stream>>>(
          omega, valsb, parts + (size_t)(k >> 4) * 64 * (B_SZ * 16 * 128));
    }
  }

  k_reduce_kn<<<256, 256, 0, stream>>>(parts, knear);
  k_ffdim<<<dim3(64, B_SZ), 256, 0, stream>>>(knear, Wd, cast1);
  k_final<<<dim3(128, B_SZ), 128, 0, stream>>>(x, cast1, Wc, gamma, beta, out);
}
